// Seq2SeqDecode_39694087750036
// MI455X (gfx1250) — compile-verified, matrix operand order checked by reading, not run on hardware
//
#include <hip/hip_runtime.h>

// Problem constants (from the reference)
#define BB 32
#define TT 64
#define VV 32000
#define EE 512
#define HH 1024
#define GG 4096   // 4*H
#define MM 2048   // B*T

#define GEMM_MBLK 256   // 8 waves x 2 M-tiles x 16 rows per block
#define GEMM_NBLK 64    // 4 N-tiles of 16 per wave
#define KSTEP 32

#define LSTM_WAVES 4
#define LSTM_KSLICE (HH / LSTM_WAVES)   // 256

typedef __attribute__((ext_vector_type(16))) __bf16 v16bf;
typedef __attribute__((ext_vector_type(8)))  float  v8f;
typedef __attribute__((ext_vector_type(4)))  unsigned int v4u;
typedef __attribute__((ext_vector_type(8)))  int v8i;
typedef __attribute__((ext_vector_type(4)))  int v4i;

// ---- Tensor Data Mover availability / arity discrimination -----------------
#if defined(__has_builtin)
#  if __has_builtin(__builtin_amdgcn_tensor_load_to_lds)
#    define HAVE_TDM 1
#  endif
#endif
#ifndef HAVE_TDM
#  define HAVE_TDM 0
#endif
#if __has_include(<hip/amd_detail/amd_gfx1250_TDM.h>)
#  define TDM_6ARG 1   // amdgpu-toolchain (clang-23 / therock headers): 6-arg builtin
#endif

union FragU { v16bf v; unsigned u[8]; };

// fp32 -> bf16 round-to-nearest-even
__device__ __forceinline__ unsigned short f2bf(float f) {
  unsigned u = __float_as_uint(f);
  u += 0x7fffu + ((u >> 16) & 1u);
  return (unsigned short)(u >> 16);
}

// 16-bit WMMA fragment load from row-major bf16 with leading dim ld.
// CDNA5 16-bit A layout: lane L -> row, half = L>>4,
// VGPR v holds packed K pair: k = (v/4)*16 + half*8 + (v%4)*2.
// Collapses to two b128 loads per lane (offsets 0 and 32 bytes).
__device__ __forceinline__ v16bf load_frag16(const unsigned short* __restrict__ base,
                                             long row, int ld, int k0, int half) {
  FragU f;
  const unsigned short* p = base + row * (long)ld + k0;
#pragma unroll
  for (int v = 0; v < 8; ++v) {
    int k = ((v >> 2) << 4) + (half << 3) + ((v & 3) << 1);
    f.u[v] = *(const unsigned*)(p + k);
  }
  return f.v;
}

__device__ __forceinline__ v8f wmma_bf16(v16bf a, v16bf b, v8f c) {
  return __builtin_amdgcn_wmma_f32_16x16x32_bf16(false, a, false, b, (short)0, c,
                                                 false, false);
}

__device__ __forceinline__ float sigmoidf_(float x) {
  return 1.0f / (1.0f + __expf(-x));
}

#if HAVE_TDM
// Issue a TDM load of a 64-row x 32-elem bf16 panel (rows stride ldK elements)
// into LDS at lds_byte_addr. D# packing per CDNA5 ISA ch.8:
//  group0: [1:0]=count=1, [63:32]=lds_addr, [120:64]=global_addr, [127:126]=type=2
//  group1: data_size=1(2B), tensor_dim0=32, tensor_dim1=64, tile 32x64,
//          tensor_dim0_stride=ldK
__device__ __forceinline__ void tdm_issue_panel(unsigned lds_byte_addr,
                                                const unsigned short* gsrc,
                                                int ldK) {
  unsigned long long ga = (unsigned long long)(uintptr_t)gsrc;
  v4u g0;
  g0[0] = 1u;                                           // count=1, user mode
  g0[1] = lds_byte_addr;                                // lds_addr
  g0[2] = (unsigned)(ga & 0xffffffffu);                 // global_addr[31:0]
  g0[3] = (unsigned)((ga >> 32) & 0x01ffffffu) | (2u << 30); // addr[56:32]|type=2

  const unsigned td0 = 32u, td1 = 64u, tile0 = 32u, tile1 = 64u;
  unsigned long long s0 = (unsigned long long)(unsigned)ldK;  // dim0 stride
  v8i g1;
  g1[0] = (int)(1u << 16);                              // data_size = 2 bytes
  g1[1] = (int)((td0 & 0xffffu) << 16);                 // tensor_dim0[15:0]
  g1[2] = (int)(((td0 >> 16) & 0xffffu) | ((td1 & 0xffffu) << 16));
  g1[3] = (int)(((td1 >> 16) & 0xffffu) | ((tile0 & 0xffffu) << 16));
  g1[4] = (int)(tile1 & 0xffffu);                       // tile_dim1, tile_dim2=0
  g1[5] = (int)(unsigned)(s0 & 0xffffffffu);            // dim0_stride[31:0]
  g1[6] = (int)(unsigned)((s0 >> 32) & 0xffffu);        // dim0_stride[47:32]
  g1[7] = 0;

  v4i z4 = {0, 0, 0, 0};
#ifdef TDM_6ARG
  v8i z8 = {0, 0, 0, 0, 0, 0, 0, 0};
  __builtin_amdgcn_tensor_load_to_lds(g0, g1, z4, z4, z8, 0);
#else
  __builtin_amdgcn_tensor_load_to_lds(g0, g1, z4, z4, 0);
#endif
}
#endif  // HAVE_TDM

// Fallback: cooperative synchronous staging of a 64x32 bf16 panel.
__device__ __forceinline__ void stage_panel_sync(unsigned short* dst,
                                                 const unsigned short* src,
                                                 int ldK) {
  const int t   = threadIdx.x;     // 256 threads
  const int row = t >> 2;          // 4 threads per row
  const int kk  = (t & 3) * 8;     // 8 bf16 each
  const unsigned* s = (const unsigned*)(src + (long)row * ldK + kk);
  unsigned* d = (unsigned*)(dst + row * KSTEP + kk);
  d[0] = s[0]; d[1] = s[1]; d[2] = s[2]; d[3] = s[3];
}

// ---------------------------------------------------------------------------
// Prologue kernels
// ---------------------------------------------------------------------------
__global__ void conv_transpose_bf16(const float* __restrict__ src,    // [K,N] f32
                                    unsigned short* __restrict__ dst, // [N,K] bf16
                                    int K, int N) {
  size_t total = (size_t)K * (size_t)N;
  size_t stride = (size_t)gridDim.x * blockDim.x;
  for (size_t i = (size_t)blockIdx.x * blockDim.x + threadIdx.x; i < total; i += stride) {
    size_t n = i / (size_t)K;
    size_t k = i - n * (size_t)K;
    dst[i] = f2bf(src[k * (size_t)N + n]);
  }
}

__global__ void gather_emb_bf16(const int* __restrict__ x,
                                const float* __restrict__ table,
                                unsigned short* __restrict__ out) {
  size_t total = (size_t)MM * EE;
  size_t stride = (size_t)gridDim.x * blockDim.x;
  for (size_t i = (size_t)blockIdx.x * blockDim.x + threadIdx.x; i < total; i += stride) {
    int m = (int)(i >> 9);   // E = 512
    int e = (int)(i & 511);
    out[i] = f2bf(table[(size_t)x[m] * EE + e]);
  }
}

__global__ void init_state(const float* __restrict__ state_h,
                           const float* __restrict__ state_c,
                           unsigned short* __restrict__ h_bf,
                           float* __restrict__ c_ws) {
  int total = BB * HH;
  for (int i = blockIdx.x * blockDim.x + threadIdx.x; i < total;
       i += gridDim.x * blockDim.x) {
    h_bf[i] = f2bf(state_h[i]);
    c_ws[i] = state_c[i];
  }
}

// ---------------------------------------------------------------------------
// Block-cooperative WMMA GEMM with TDM-staged B panels.
//   C[M,N] = A[M,K](bf16) @ Bt[N,K]^T(bf16) + bias[N]  (fp32 out)
// Block = 256 threads = 8 waves; each wave owns 2 M-tiles x 4 N-tiles
// (block output tile 256x64). B panel (64 x 32, 4KB) double-buffered in LDS
// via the Tensor Data Mover; A fragments register double-buffered.
// grid: (M/256, N/64)
// ---------------------------------------------------------------------------
__global__ __launch_bounds__(256) void wmma_gemm_bias(
    const unsigned short* __restrict__ A,
    const unsigned short* __restrict__ Bt,
    const float* __restrict__ bias,
    float* __restrict__ C,
    int N, int K) {
  __shared__ __align__(16) unsigned short sB[2][GEMM_NBLK * KSTEP];  // 2 x 4KB

  const int tid  = threadIdx.x;
  const int wid  = tid >> 5;
  const int lane = tid & 31;
  const int half = lane >> 4;
  const int nr   = lane & 15;
  const long m0  = (long)blockIdx.x * GEMM_MBLK + (long)wid * 32;  // 2 M-tiles
  const int  n0  = blockIdx.y * GEMM_NBLK;
  const int  nsteps = K / KSTEP;
  const unsigned short* Bpanel = Bt + (long)n0 * K;

#if HAVE_TDM
  // Preload panel 0 via the Tensor Data Mover (wave 0 issues; TENSORcnt)
  if (tid < 32) {
    tdm_issue_panel((unsigned)(uintptr_t)(void*)&sB[0][0], Bpanel, K);
    __builtin_amdgcn_s_wait_tensorcnt(0);
  }
  __syncthreads();
#endif

  v8f acc[2][4];
#pragma unroll
  for (int mt = 0; mt < 2; ++mt)
#pragma unroll
    for (int j = 0; j < 4; ++j) acc[mt][j] = (v8f){};

  v16bf aCur0 = load_frag16(A, m0 + nr,      K, 0, half);
  v16bf aCur1 = load_frag16(A, m0 + 16 + nr, K, 0, half);

#pragma unroll 1
  for (int i = 0; i < nsteps; ++i) {
    const int cur = i & 1;

#if HAVE_TDM
    // Kick off async DMA of the next panel while we compute on this one.
    if (i + 1 < nsteps && tid < 32)
      tdm_issue_panel((unsigned)(uintptr_t)(void*)&sB[cur ^ 1][0],
                      Bpanel + (i + 1) * KSTEP, K);
#else
    __syncthreads();
    stage_panel_sync(&sB[cur][0], Bpanel + i * KSTEP, K);
    __syncthreads();
#endif

    // Register double-buffer the two A fragments.
    v16bf aNxt0 = aCur0, aNxt1 = aCur1;
    if (i + 1 < nsteps) {
      aNxt0 = load_frag16(A, m0 + nr,      K, (i + 1) * KSTEP, half);
      aNxt1 = load_frag16(A, m0 + 16 + nr, K, (i + 1) * KSTEP, half);
      if (i + 2 < nsteps)
        __builtin_prefetch(A + (m0 + nr) * (long)K + (i + 2) * KSTEP, 0, 1);
    }

    const unsigned short* pb = &sB[cur][0];
#pragma unroll
    for (int j = 0; j < 4; ++j) {
      v16bf bf = load_frag16(pb, (long)(j * 16 + nr), KSTEP, 0, half);
      acc[0][j] = wmma_bf16(aCur0, bf, acc[0][j]);
      acc[1][j] = wmma_bf16(aCur1, bf, acc[1][j]);
    }

#if HAVE_TDM
    if (i + 1 < nsteps) {
      if (tid < 32) __builtin_amdgcn_s_wait_tensorcnt(0);
      __syncthreads();
    }
#endif
    aCur0 = aNxt0;
    aCur1 = aNxt1;
  }

#pragma unroll
  for (int mt = 0; mt < 2; ++mt)
#pragma unroll
    for (int r = 0; r < 8; ++r) {
      long m = m0 + mt * 16 + r + (half << 3);  // C layout: lanes16-31 -> M=r+8
      float* crow = C + m * (long)N;
      int c0 = n0 + nr;
      crow[c0]      = acc[mt][0][r] + bias[c0];
      crow[c0 + 16] = acc[mt][1][r] + bias[c0 + 16];
      crow[c0 + 32] = acc[mt][2][r] + bias[c0 + 32];
      crow[c0 + 48] = acc[mt][3][r] + bias[c0 + 48];
    }
}

// ---------------------------------------------------------------------------
// One LSTM timestep, K split 4 ways across 4 waves per block.
// Each block owns 16 H-columns of all 4 gates. Wave w accumulates the
// K-slice [w*256, w*256+256) into register tiles (double-buffered frags),
// partials go to LDS, then a block-wide reduction applies the gate math.
// grid: (H/16) blocks, 128 threads.
// ---------------------------------------------------------------------------
__global__ __launch_bounds__(128) void lstm_step(
    const float* __restrict__ xW,            // [B*T,4H] f32
    const unsigned short* __restrict__ Ut,   // [4H,H] bf16
    const unsigned short* __restrict__ h_src,// [B,H] bf16
    unsigned short* __restrict__ h_dst,      // [B,H] bf16
    float* __restrict__ c_state,             // [B,H] f32
    unsigned short* __restrict__ seq_bf,     // [B*T,H] bf16
    float* __restrict__ outH,
    float* __restrict__ outC,
    int t, int last) {
  // part[w][g][bb][col] : 4*4*32*16 floats = 32 KB
  __shared__ __align__(16) float part[LSTM_WAVES * 4 * BB * 16];

  const int tid  = threadIdx.x;
  const int wid  = tid >> 5;
  const int lane = tid & 31;
  const int half = lane >> 4;
  const int nr   = lane & 15;
  const int c0   = blockIdx.x * 16;
  const int kb   = wid * LSTM_KSLICE;

  v8f acc[4][2];
#pragma unroll
  for (int g = 0; g < 4; ++g)
#pragma unroll
    for (int mt = 0; mt < 2; ++mt) acc[g][mt] = (v8f){};

  v16bf a0c = load_frag16(h_src, (long)nr,        HH, kb, half);
  v16bf a1c = load_frag16(h_src, (long)(16 + nr), HH, kb, half);
  v16bf bc[4];
#pragma unroll
  for (int g = 0; g < 4; ++g)
    bc[g] = load_frag16(Ut, (long)g * HH + c0 + nr, HH, kb, half);

#pragma unroll 1
  for (int k0 = kb; k0 < kb + LSTM_KSLICE - KSTEP; k0 += KSTEP) {
    const int kn = k0 + KSTEP;
    v16bf a0n = load_frag16(h_src, (long)nr,        HH, kn, half);
    v16bf a1n = load_frag16(h_src, (long)(16 + nr), HH, kn, half);
    v16bf bn[4];
#pragma unroll
    for (int g = 0; g < 4; ++g)
      bn[g] = load_frag16(Ut, (long)g * HH + c0 + nr, HH, kn, half);

#pragma unroll
    for (int g = 0; g < 4; ++g) {
      acc[g][0] = wmma_bf16(a0c, bc[g], acc[g][0]);
      acc[g][1] = wmma_bf16(a1c, bc[g], acc[g][1]);
    }
    a0c = a0n; a1c = a1n;
#pragma unroll
    for (int g = 0; g < 4; ++g) bc[g] = bn[g];
  }
  // Peeled last K-step of this slice
#pragma unroll
  for (int g = 0; g < 4; ++g) {
    acc[g][0] = wmma_bf16(a0c, bc[g], acc[g][0]);
    acc[g][1] = wmma_bf16(a1c, bc[g], acc[g][1]);
  }

  // Spill partial sums to LDS
  float* myp = &part[wid * (4 * BB * 16)];
#pragma unroll
  for (int g = 0; g < 4; ++g)
#pragma unroll
    for (int mt = 0; mt < 2; ++mt)
#pragma unroll
      for (int r = 0; r < 8; ++r) {
        int bb = mt * 16 + r + (half << 3);
        myp[g * (BB * 16) + bb * 16 + nr] = acc[g][mt][r];
      }
  __syncthreads();

  // Reduce partials + input projection, apply gates (Keras order i,f,g,o).
  // 512 (bb,col) pairs over 128 threads -> 4 each.
#pragma unroll 1
  for (int p = tid; p < BB * 16; p += 128) {
    const int bb  = p >> 4;
    const int col = c0 + (p & 15);
    float z[4];
#pragma unroll
    for (int g = 0; g < 4; ++g) {
      float s = part[g * 512 + p] + part[2048 + g * 512 + p] +
                part[4096 + g * 512 + p] + part[6144 + g * 512 + p];
      z[g] = s + xW[((size_t)bb * TT + t) * GG + (size_t)g * HH + col];
    }
    float iv = sigmoidf_(z[0]);
    float fv = sigmoidf_(z[1]);
    float gv = tanhf(z[2]);
    float ov = sigmoidf_(z[3]);
    float cold = c_state[bb * HH + col];
    float cnew = fv * cold + iv * gv;
    float hnew = ov * tanhf(cnew);
    c_state[bb * HH + col] = cnew;
    unsigned short hb = f2bf(hnew);
    h_dst[bb * HH + col] = hb;
    seq_bf[((size_t)bb * TT + t) * HH + col] = hb;
    if (last) {
      outH[bb * HH + col] = hnew;
      outC[bb * HH + col] = cnew;
    }
  }
}

// ---------------------------------------------------------------------------
// Host orchestration
// ---------------------------------------------------------------------------
extern "C" void kernel_launch(void* const* d_in, const int* in_sizes, int n_in,
                              void* d_out, int out_size, void* d_ws, size_t ws_size,
                              hipStream_t stream) {
  const int*   x       = (const int*)  d_in[0];
  const float* state_h = (const float*)d_in[1];
  const float* state_c = (const float*)d_in[2];
  const float* emb_tab = (const float*)d_in[3];
  const float* W       = (const float*)d_in[4];
  const float* U       = (const float*)d_in[5];
  const float* bvec    = (const float*)d_in[6];
  const float* dW      = (const float*)d_in[7];
  const float* db      = (const float*)d_in[8];
  float* out = (float*)d_out;

  // Workspace carving (256B aligned)
  char* ws = (char*)d_ws;
  size_t off = 0;
  auto carve = [&](size_t bytes) -> void* {
    off = (off + 255) & ~(size_t)255;
    void* p = ws + off;
    off += bytes;
    return p;
  };
  unsigned short* emb_bf = (unsigned short*)carve((size_t)MM * EE * 2);  // 2 MB
  unsigned short* Wt     = (unsigned short*)carve((size_t)GG * EE * 2);  // 4 MB
  unsigned short* Ut     = (unsigned short*)carve((size_t)GG * HH * 2);  // 8 MB
  unsigned short* DWt    = (unsigned short*)carve((size_t)VV * HH * 2);  // 62.5 MB (L2-resident)
  float*          xW     = (float*)         carve((size_t)MM * GG * 4);  // 32 MB
  unsigned short* seq_bf = (unsigned short*)carve((size_t)MM * HH * 2);  // 4 MB
  unsigned short* h_bf0  = (unsigned short*)carve((size_t)BB * HH * 2);
  unsigned short* h_bf1  = (unsigned short*)carve((size_t)BB * HH * 2);
  float*          c_ws   = (float*)         carve((size_t)BB * HH * 4);

  // 1) Weights -> bf16, transposed to [N,K]
  {
    size_t tot = (size_t)EE * GG;
    conv_transpose_bf16<<<(int)((tot + 255) / 256), 256, 0, stream>>>(W, Wt, EE, GG);
  }
  {
    size_t tot = (size_t)HH * GG;
    conv_transpose_bf16<<<(int)((tot + 255) / 256), 256, 0, stream>>>(U, Ut, HH, GG);
  }
  {
    size_t tot = (size_t)HH * VV;
    conv_transpose_bf16<<<(int)((tot + 255) / 256), 256, 0, stream>>>(dW, DWt, HH, VV);
  }

  // 2) Embedding gather -> bf16
  {
    size_t tot = (size_t)MM * EE;
    gather_emb_bf16<<<(int)((tot + 255) / 256), 256, 0, stream>>>(x, emb_tab, emb_bf);
  }

  // 3) Initial state
  init_state<<<128, 256, 0, stream>>>(state_h, state_c, h_bf0, c_ws);

  // 4) Input projection: xW[B*T,4H] = emb @ W + b
  {
    dim3 grid(MM / GEMM_MBLK, GG / GEMM_NBLK);   // 8 x 64
    wmma_gemm_bias<<<grid, 256, 0, stream>>>(emb_bf, Wt, bvec, xW, GG, EE);
  }

  // 5) LSTM scan over T timesteps (h ping-pongs between bf16 buffers)
  float* outH = out + (size_t)MM * VV;
  float* outC = outH + (size_t)BB * HH;
  for (int t = 0; t < TT; ++t) {
    const unsigned short* hsrc = (t & 1) ? h_bf1 : h_bf0;
    unsigned short*       hdst = (t & 1) ? h_bf0 : h_bf1;
    lstm_step<<<HH / 16, 128, 0, stream>>>(xW, Ut, hsrc, hdst, c_ws, seq_bf,
                                           outH, outC, t, (t == TT - 1) ? 1 : 0);
  }

  // 6) Logits: out[B*T,V] = seq @ dense_W + dense_b  (the 134-GFLOP GEMM)
  {
    dim3 grid(MM / GEMM_MBLK, VV / GEMM_NBLK);   // 8 x 500
    wmma_gemm_bias<<<grid, 256, 0, stream>>>(seq_bf, DWt, db, out, VV, HH);
  }
}
